// RadarDynamicClassifier_86406152061388
// MI455X (gfx1250) — matrix operand order repaired and checked by reading, not compiled, not run on hardware
//
#include <hip/hip_runtime.h>

// ---------------------------------------------------------------------------
// DGCNN-style pipeline for MI455X (gfx1250, wave32, WMMA 16x16x32 f16->f32).
// GEMMs (EdgeConv MLPs, classifier, conv2 Gram/kNN) on v_wmma_f32_16x16x32_f16.
// A-fragments are register-resident across all N-tiles (activations read once).
// BN training-mode stats via register accumulation + LDS combine.
// ---------------------------------------------------------------------------

#define BB    8
#define NPTS  4096
#define KNN   20
#define MROWS (BB * NPTS * KNN)   // 655360 rows through the EdgeConv MLPs
#define BNROW (BB * NPTS)         // 32768

typedef __attribute__((ext_vector_type(16))) _Float16 v16h;
typedef __attribute__((ext_vector_type(8)))  _Float16 v8h;
typedef __attribute__((ext_vector_type(8)))  float    v8f;

// ---- WMMA fragment helpers -------------------------------------------------
// A (16x32 f16, row major, lda halves):
//   lanes 0-15 : row = lane,     K = 0..15
//   lanes 16-31: row = lane-16,  K = 16..31
__device__ inline v16h load_a_frag(const _Float16* A, int lda, int m0, int k0, int lane) {
  int row  = m0 + (lane & 15);
  int koff = (lane < 16) ? 0 : 16;
  const _Float16* p = A + (size_t)row * lda + k0 + koff;
  union { v16h v; struct { v8h lo, hi; } s; } u;
  u.s.lo = *(const v8h*)p;
  u.s.hi = *(const v8h*)(p + 8);
  return u.v;
}

// B (32x16 f16): lanes 0-15 hold col=lane, K=k0..k0+15; lanes 16-31 hold
// col=lane-16, K=k0+16..k0+31.  Pre-packed weights: 32 lanes x 16 halves.
__device__ inline v16h load_b_frag_packed(const _Float16* Wp, int lane) {
  const _Float16* p = Wp + (size_t)lane * 16;
  union { v16h v; struct { v8h lo, hi; } s; } u;
  u.s.lo = *(const v8h*)p;
  u.s.hi = *(const v8h*)(p + 8);
  return u.v;
}

// B-frag where B = X^T (Gram): column n of B is row n of X, so each lane
// reads 16 contiguous halves of one X row.
__device__ inline v16h load_bt_frag(const _Float16* X, int ldx, int c0, int k0, int lane) {
  int col  = c0 + (lane & 15);
  int koff = (lane < 16) ? 0 : 16;
  const _Float16* p = X + (size_t)col * ldx + k0 + koff;
  union { v16h v; struct { v8h lo, hi; } s; } u;
  u.s.lo = *(const v8h*)p;
  u.s.hi = *(const v8h*)(p + 8);
  return u.v;
}

// ---- Weight pre-pack: W[Korig x Nn] f32 -> f16 B-fragments, K padded to Kp.
__global__ void pack_w_kernel(const float* __restrict__ W, _Float16* __restrict__ Wp,
                              int Kp, int Nn, int Korig) {
  int t = blockIdx.x * blockDim.x + threadIdx.x;
  int nkt = Kp / 32, nnt = Nn / 16;
  if (t >= nnt * nkt * 32) return;
  int lane = t & 31, tile = t >> 5;
  int tk = tile % nkt, tn = tile / nkt;
  int col   = tn * 16 + (lane & 15);
  int kbase = tk * 32 + ((lane < 16) ? 0 : 16);
  _Float16* o = Wp + (size_t)t * 16;
#pragma unroll
  for (int kk = 0; kk < 16; ++kk) {
    int k = kbase + kk;
    o[kk] = (_Float16)((k < Korig) ? W[(size_t)k * Nn + col] : 0.f);
  }
}

// ---- GEMM: C[M,NN] f32 = A[M,KP] f16 * Wp(packed) + bias.
// 4 waves/block; each wave owns a 16-row strip, keeps ALL A K-fragments in
// registers and sweeps the N-tiles reusing them (activations read once).
template <int KP, int NN>
__global__ void gemm_f16_tmpl(const _Float16* __restrict__ A, const _Float16* __restrict__ Wp,
                              const float* __restrict__ bias, float* __restrict__ C, int M) {
  constexpr int NKT = KP / 32, NNT = NN / 16;
  int lane = threadIdx.x & 31, wave = threadIdx.x >> 5;
  int m0 = (blockIdx.x * 4 + wave) * 16;
  if (m0 >= M) return;

  v16h a[NKT];
#pragma unroll
  for (int tk = 0; tk < NKT; ++tk) a[tk] = load_a_frag(A, KP, m0, tk * 32, lane);

  int colr  = lane & 15;
  int rbase = m0 + ((lane < 16) ? 0 : 8);
#pragma unroll
  for (int nt = 0; nt < NNT; ++nt) {
    v8f acc = {};
#pragma unroll
    for (int tk = 0; tk < NKT; ++tk) {
      v16h b = load_b_frag_packed(Wp + ((size_t)(nt * NKT + tk) * 32) * 16, lane);
      acc = __builtin_amdgcn_wmma_f32_16x16x32_f16(false, a[tk], false, b, (short)0, acc, false, false);
    }
    int col = nt * 16 + colr;
    float bv = bias[col];
#pragma unroll
    for (int v = 0; v < 8; ++v)
      C[(size_t)(rbase + v) * NN + col] = acc[v] + bv;
  }
}

// ---- kNN on raw input (C=4): one thread per point, insertion top-20 -------
__global__ void knn_c4_kernel(const float* __restrict__ x, int* __restrict__ idx_out) {
  int g = blockIdx.x * blockDim.x + threadIdx.x;
  if (g >= BNROW) return;
  int b = g / NPTS;
  const float4* xb = (const float4*)(x + (size_t)b * NPTS * 4);
  float4 xi = ((const float4*)x)[g];
  float bd[KNN]; int bi[KNN];
#pragma unroll
  for (int j = 0; j < KNN; ++j) { bd[j] = 3.4e38f; bi[j] = 0; }
  for (int m = 0; m < NPTS; ++m) {
    float4 xm = xb[m];
    float d0 = xm.x - xi.x, d1 = xm.y - xi.y, d2 = xm.z - xi.z, d3 = xm.w - xi.w;
    float d = d0 * d0 + d1 * d1 + d2 * d2 + d3 * d3;
    if (d < bd[KNN - 1]) {
      int p = KNN - 1;
      while (p > 0 && bd[p - 1] > d) { bd[p] = bd[p - 1]; bi[p] = bi[p - 1]; --p; }
      bd[p] = d; bi[p] = m;
    }
  }
  for (int j = 0; j < KNN; ++j) idx_out[(size_t)g * KNN + j] = bi[j];
}

// ---- msg builders ----------------------------------------------------------
__global__ void build_msg1_kernel(const float* __restrict__ x, const int* __restrict__ idx,
                                  _Float16* __restrict__ msg) {
  int r = blockIdx.x * blockDim.x + threadIdx.x;
  if (r >= MROWS) return;
  int g = r / KNN, b = g / NPTS;
  const float* xi = x + (size_t)g * 4;
  const float* xj = x + ((size_t)b * NPTS + idx[r]) * 4;
  _Float16* o = msg + (size_t)r * 32;
#pragma unroll
  for (int c = 0; c < 4; ++c) {
    o[c]     = (_Float16)xi[c];
    o[4 + c] = (_Float16)(xj[c] - xi[c]);
  }
#pragma unroll
  for (int c = 8; c < 32; ++c) o[c] = (_Float16)0.f;  // K pad 8 -> 32
}

__global__ void build_msg2_kernel(const float* __restrict__ x1, const int* __restrict__ idx,
                                  _Float16* __restrict__ msg) {
  int r = blockIdx.x * blockDim.x + threadIdx.x;
  if (r >= MROWS) return;
  int g = r / KNN, b = g / NPTS;
  const float* xi = x1 + (size_t)g * 64;
  const float* xj = x1 + ((size_t)b * NPTS + idx[r]) * 64;
  _Float16* o = msg + (size_t)r * 128;
#pragma unroll 8
  for (int c = 0; c < 64; ++c) {
    float a = xi[c];
    o[c]      = (_Float16)a;
    o[64 + c] = (_Float16)(xj[c] - a);
  }
}

__global__ void build_cat_kernel(const float* __restrict__ x1, const float* __restrict__ x2,
                                 _Float16* __restrict__ cat) {
  long long i = (long long)blockIdx.x * blockDim.x + threadIdx.x;
  if (i >= (long long)BNROW * 192) return;
  int c = (int)(i % 192); long long bn = i / 192;
  float v = (c < 64) ? x1[bn * 64 + c] : x2[bn * 128 + (c - 64)];
  cat[i] = (_Float16)v;
}

// ---- BatchNorm stats (training mode): sum & sumsq per channel -------------
// Thread t owns channel (t & (C-1)); register accumulation over rows, then
// LDS combine across row-groups, then 2 global atomics per channel per block.
__global__ void zero_kernel(float* p, int n) {
  int i = blockIdx.x * blockDim.x + threadIdx.x;
  if (i < n) p[i] = 0.f;
}

template <int C>
__global__ void bn_stats_tmpl(const float* __restrict__ pre, float* __restrict__ stats,
                              long long M) {
  constexpr int RPB = 256 / C;            // row-groups per block
  int c  = threadIdx.x & (C - 1);
  int rg = threadIdx.x / C;
  float s = 0.f, ss = 0.f;
  long long stride = (long long)gridDim.x * RPB;
  for (long long r = (long long)blockIdx.x * RPB + rg; r < M; r += stride) {
    float v = pre[r * C + c];
    s += v; ss += v * v;
  }
  __shared__ float ls[512];
  ls[threadIdx.x]       = s;
  ls[256 + threadIdx.x] = ss;
  __syncthreads();
  if (rg == 0) {
#pragma unroll
    for (int g = 1; g < RPB; ++g) {
      s  += ls[g * C + c];
      ss += ls[256 + g * C + c];
    }
    atomicAdd(&stats[c], s);
    atomicAdd(&stats[C + c], ss);
  }
}

// ---- BN + ReLU (f16 out for next GEMM, optional f32 out) ------------------
__global__ void bn_relu_kernel(const float* __restrict__ pre, const float* __restrict__ stats,
                               const float* __restrict__ g, const float* __restrict__ be,
                               _Float16* __restrict__ out16, float* __restrict__ out32,
                               long long M, int C, float invM) {
  long long i = (long long)blockIdx.x * blockDim.x + threadIdx.x;
  if (i >= M * C) return;
  int c = (int)(i & (C - 1));
  float mean = stats[c] * invM;
  float var  = stats[C + c] * invM - mean * mean;
  float sc = g[c] * rsqrtf(var + 1e-5f);
  float sh = be[c] - mean * sc;
  float h = fmaxf(pre[i] * sc + sh, 0.f);
  out16[i] = (_Float16)h;
  if (out32) out32[i] = h;
}

// ---- BN + ReLU + max over k neighbors -> [BN, C] (f32 + f16) --------------
__global__ void bn_relu_kmax_kernel(const float* __restrict__ pre, const float* __restrict__ stats,
                                    const float* __restrict__ g, const float* __restrict__ be,
                                    float* __restrict__ xout, _Float16* __restrict__ xhout, int C,
                                    float invM) {
  long long i = (long long)blockIdx.x * blockDim.x + threadIdx.x;
  if (i >= (long long)BNROW * C) return;
  int c = (int)(i % C);
  long long bn = i / C;
  float mean = stats[c] * invM;
  float var  = stats[C + c] * invM - mean * mean;
  float sc = g[c] * rsqrtf(var + 1e-5f);
  float sh = be[c] - mean * sc;
  float mx = 0.f;                        // relu output >= 0
  const float* p = pre + (bn * KNN) * (long long)C + c;
#pragma unroll
  for (int j = 0; j < KNN; ++j)
    mx = fmaxf(mx, fmaxf(p[(size_t)j * C] * sc + sh, 0.f));
  xout[i] = mx;
  if (xhout) xhout[i] = (_Float16)mx;
}

// ---- squared norms of x1 rows ---------------------------------------------
__global__ void sqnorm64_kernel(const float* __restrict__ x1, float* __restrict__ sq) {
  int g = blockIdx.x * blockDim.x + threadIdx.x;
  if (g >= BNROW) return;
  const float* r = x1 + (size_t)g * 64;
  float s = 0.f;
#pragma unroll 8
  for (int c = 0; c < 64; ++c) s += r[c] * r[c];
  sq[g] = s;
}

// ---- Gram-matrix kNN for conv2 (C=64) via WMMA -----------------------------
// Block = 128 threads (4 waves) handles 16 query rows of one batch.
// Phase 1: WMMA produces a 16x1024 distance strip in LDS.
// Phase 2: all 128 threads select: thread (row, slice-of-8) keeps a private
// top-20 over its 128-column slice; per-row leader merges 8x20 candidates.
#define CTILE 1024
__global__ void gram_knn_kernel(const _Float16* __restrict__ x1h, const float* __restrict__ sq,
                                int* __restrict__ idx_out) {
  __shared__ float dtile[16 * CTILE];
  __shared__ float cd[128 * KNN];
  __shared__ int   ci[128 * KNN];
  int tb = blockIdx.x;
  int b  = tb / (NPTS / 16);
  int r0 = (tb % (NPTS / 16)) * 16;
  int lane = threadIdx.x & 31, wave = threadIdx.x >> 5;
  size_t base = (size_t)b * NPTS;
  const _Float16* Xb = x1h + base * 64;

  v16h a0 = load_a_frag(Xb, 64, r0, 0, lane);
  v16h a1 = load_a_frag(Xb, 64, r0, 32, lane);
  int rloc = (lane < 16) ? 0 : 8;

  int myrow   = threadIdx.x >> 3;       // 0..15
  int myslice = threadIdx.x & 7;        // 0..7
  float bd[KNN]; int bi[KNN];
#pragma unroll
  for (int j = 0; j < KNN; ++j) { bd[j] = 3.4e38f; bi[j] = 0; }

  for (int ct = 0; ct < NPTS / CTILE; ++ct) {
    for (int t = wave; t < CTILE / 16; t += 4) {
      int c0 = ct * CTILE + t * 16;
      v8f acc = {};
      v16h b0 = load_bt_frag(Xb, 64, c0, 0, lane);
      acc = __builtin_amdgcn_wmma_f32_16x16x32_f16(false, a0, false, b0, (short)0, acc, false, false);
      v16h b1 = load_bt_frag(Xb, 64, c0, 32, lane);
      acc = __builtin_amdgcn_wmma_f32_16x16x32_f16(false, a1, false, b1, (short)0, acc, false, false);
      int col = c0 + (lane & 15);
      float sc = sq[base + col];
#pragma unroll
      for (int v = 0; v < 8; ++v) {
        float d = sq[base + r0 + rloc + v] + sc - 2.f * acc[v];
        dtile[(rloc + v) * CTILE + (t * 16 + (lane & 15))] = d;
      }
    }
    __syncthreads();
    {
      const float* drow = &dtile[myrow * CTILE + myslice * (CTILE / 8)];
      int gbase = ct * CTILE + myslice * (CTILE / 8);
      for (int j = 0; j < CTILE / 8; ++j) {
        float d = drow[j];
        if (d < bd[KNN - 1]) {
          int p = KNN - 1;
          while (p > 0 && bd[p - 1] > d) { bd[p] = bd[p - 1]; bi[p] = bi[p - 1]; --p; }
          bd[p] = d; bi[p] = gbase + j;
        }
      }
    }
    __syncthreads();
  }
  // stage candidates and merge per row
#pragma unroll
  for (int j = 0; j < KNN; ++j) {
    cd[threadIdx.x * KNN + j] = bd[j];
    ci[threadIdx.x * KNN + j] = bi[j];
  }
  __syncthreads();
  if (myslice == 0) {
    float fd[KNN]; int fi[KNN];
#pragma unroll
    for (int j = 0; j < KNN; ++j) { fd[j] = 3.4e38f; fi[j] = 0; }
    for (int s = 0; s < 8; ++s) {
      const float* sd = &cd[(myrow * 8 + s) * KNN];
      const int*   si = &ci[(myrow * 8 + s) * KNN];
      for (int j = 0; j < KNN; ++j) {
        float d = sd[j];
        if (d < fd[KNN - 1]) {
          int p = KNN - 1;
          // stable on ties, prefer lower index (earlier slice / earlier col)
          while (p > 0 && (fd[p - 1] > d || (fd[p - 1] == d && fi[p - 1] > si[j]))) {
            fd[p] = fd[p - 1]; fi[p] = fi[p - 1]; --p;
          }
          fd[p] = d; fi[p] = si[j];
        }
      }
    }
    int r = r0 + myrow;
    for (int j = 0; j < KNN; ++j)
      idx_out[(base + r) * KNN + j] = fi[j];
  }
}

// ---- final Linear(128 -> 1) -----------------------------------------------
__global__ void final_linear_kernel(const float* __restrict__ h, const float* __restrict__ w,
                                    const float* __restrict__ b, float* __restrict__ out) {
  int g = blockIdx.x * blockDim.x + threadIdx.x;
  if (g >= BNROW) return;
  const float* hr = h + (size_t)g * 128;
  float s = 0.f;
#pragma unroll 8
  for (int c = 0; c < 128; ++c) s += hr[c] * w[c];
  out[g] = s + b[0];
}

// ---------------------------------------------------------------------------
static inline void* bump(char*& p, size_t bytes) {
  void* r = p;
  p += (bytes + 255) & ~(size_t)255;
  return r;
}

extern "C" void kernel_launch(void* const* d_in, const int* in_sizes, int n_in,
                              void* d_out, int out_size, void* d_ws, size_t ws_size,
                              hipStream_t stream) {
  (void)in_sizes; (void)n_in; (void)out_size; (void)ws_size;
  const float* x      = (const float*)d_in[0];
  const float* c1_w1  = (const float*)d_in[1];
  const float* c1_b1  = (const float*)d_in[2];
  const float* c1_g1  = (const float*)d_in[3];
  const float* c1_be1 = (const float*)d_in[4];
  const float* c1_w2  = (const float*)d_in[5];
  const float* c1_b2  = (const float*)d_in[6];
  const float* c1_g2  = (const float*)d_in[7];
  const float* c1_be2 = (const float*)d_in[8];
  const float* c2_w1  = (const float*)d_in[9];
  const float* c2_b1  = (const float*)d_in[10];
  const float* c2_g1  = (const float*)d_in[11];
  const float* c2_be1 = (const float*)d_in[12];
  const float* c2_w2  = (const float*)d_in[13];
  const float* c2_b2  = (const float*)d_in[14];
  const float* c2_g2  = (const float*)d_in[15];
  const float* c2_be2 = (const float*)d_in[16];
  const float* clsw1  = (const float*)d_in[17];
  const float* clsb1  = (const float*)d_in[18];
  const float* clsg1  = (const float*)d_in[19];
  const float* clsbe1 = (const float*)d_in[20];
  const float* clsw2  = (const float*)d_in[21];
  const float* clsb2  = (const float*)d_in[22];
  float* out = (float*)d_out;

  char* wp = (char*)d_ws;
  int*       idx1  = (int*)      bump(wp, (size_t)MROWS * sizeof(int));
  int*       idx2  = (int*)      bump(wp, (size_t)MROWS * sizeof(int));
  _Float16*  msg16 = (_Float16*) bump(wp, (size_t)MROWS * 128 * sizeof(_Float16));
  float*     pre   = (float*)    bump(wp, (size_t)MROWS * 128 * sizeof(float));
  _Float16*  act16 = (_Float16*) bump(wp, (size_t)MROWS * 128 * sizeof(_Float16));
  float*     x1f   = (float*)    bump(wp, (size_t)BNROW * 64 * sizeof(float));
  _Float16*  x1h   = (_Float16*) bump(wp, (size_t)BNROW * 64 * sizeof(_Float16));
  float*     x2f   = (float*)    bump(wp, (size_t)BNROW * 128 * sizeof(float));
  float*     sq1   = (float*)    bump(wp, (size_t)BNROW * sizeof(float));
  _Float16*  cat16 = (_Float16*) bump(wp, (size_t)BNROW * 192 * sizeof(_Float16));
  float*     hcls  = (float*)    bump(wp, (size_t)BNROW * 128 * sizeof(float));
  float*     stats = (float*)    bump(wp, 256 * sizeof(float));
  _Float16*  wpack = (_Float16*) bump(wp, 65536 * sizeof(_Float16));

  const float invM1 = 1.f / (float)MROWS;
  const float invBN = 1.f / (float)BNROW;
  auto g1 = [](long long n) { return dim3((unsigned)((n + 255) / 256)); };

  // ---------------- EdgeConv 1 ----------------
  knn_c4_kernel<<<g1(BNROW), 256, 0, stream>>>(x, idx1);
  build_msg1_kernel<<<g1(MROWS), 256, 0, stream>>>(x, idx1, msg16);

  pack_w_kernel<<<g1((32/32)*(64/16)*32), 256, 0, stream>>>(c1_w1, wpack, 32, 64, 8);
  gemm_f16_tmpl<32, 64><<<MROWS/64, 128, 0, stream>>>(msg16, wpack, c1_b1, pre, MROWS);
  zero_kernel<<<1, 256, 0, stream>>>(stats, 256);
  bn_stats_tmpl<64><<<1024, 256, 0, stream>>>(pre, stats, MROWS);
  bn_relu_kernel<<<g1((long long)MROWS*64), 256, 0, stream>>>(pre, stats, c1_g1, c1_be1, act16, nullptr, MROWS, 64, invM1);

  pack_w_kernel<<<g1((64/32)*(64/16)*32), 256, 0, stream>>>(c1_w2, wpack, 64, 64, 64);
  gemm_f16_tmpl<64, 64><<<MROWS/64, 128, 0, stream>>>(act16, wpack, c1_b2, pre, MROWS);
  zero_kernel<<<1, 256, 0, stream>>>(stats, 256);
  bn_stats_tmpl<64><<<1024, 256, 0, stream>>>(pre, stats, MROWS);
  bn_relu_kmax_kernel<<<g1((long long)BNROW*64), 256, 0, stream>>>(pre, stats, c1_g2, c1_be2, x1f, x1h, 64, invM1);

  // ---------------- EdgeConv 2 ----------------
  sqnorm64_kernel<<<g1(BNROW), 256, 0, stream>>>(x1f, sq1);
  gram_knn_kernel<<<BB * (NPTS / 16), 128, 0, stream>>>(x1h, sq1, idx2);
  build_msg2_kernel<<<g1(MROWS), 256, 0, stream>>>(x1f, idx2, msg16);

  pack_w_kernel<<<g1((128/32)*(128/16)*32), 256, 0, stream>>>(c2_w1, wpack, 128, 128, 128);
  gemm_f16_tmpl<128, 128><<<MROWS/64, 128, 0, stream>>>(msg16, wpack, c2_b1, pre, MROWS);
  zero_kernel<<<1, 256, 0, stream>>>(stats, 256);
  bn_stats_tmpl<128><<<1024, 256, 0, stream>>>(pre, stats, MROWS);
  bn_relu_kernel<<<g1((long long)MROWS*128), 256, 0, stream>>>(pre, stats, c2_g1, c2_be1, act16, nullptr, MROWS, 128, invM1);

  pack_w_kernel<<<g1((128/32)*(128/16)*32), 256, 0, stream>>>(c2_w2, wpack, 128, 128, 128);
  gemm_f16_tmpl<128, 128><<<MROWS/64, 128, 0, stream>>>(act16, wpack, c2_b2, pre, MROWS);
  zero_kernel<<<1, 256, 0, stream>>>(stats, 256);
  bn_stats_tmpl<128><<<1024, 256, 0, stream>>>(pre, stats, MROWS);
  bn_relu_kmax_kernel<<<g1((long long)BNROW*128), 256, 0, stream>>>(pre, stats, c2_g2, c2_be2, x2f, nullptr, 128, invM1);

  // ---------------- Classifier ----------------
  build_cat_kernel<<<g1((long long)BNROW*192), 256, 0, stream>>>(x1f, x2f, cat16);
  pack_w_kernel<<<g1((192/32)*(128/16)*32), 256, 0, stream>>>(clsw1, wpack, 192, 128, 192);
  gemm_f16_tmpl<192, 128><<<BNROW/64, 128, 0, stream>>>(cat16, wpack, clsb1, pre, BNROW);
  zero_kernel<<<1, 256, 0, stream>>>(stats, 256);
  bn_stats_tmpl<128><<<1024, 256, 0, stream>>>(pre, stats, BNROW);
  bn_relu_kernel<<<g1((long long)BNROW*128), 256, 0, stream>>>(pre, stats, clsg1, clsbe1, act16, hcls, BNROW, 128, invBN);

  final_linear_kernel<<<g1(BNROW), 256, 0, stream>>>(hcls, clsw2, clsb2, out);
}
